// ResGAT_25494925869738
// MI455X (gfx1250) — compile-verified, matrix operand order checked by reading, not run on hardware
//
#include <hip/hip_runtime.h>
#include <hip/hip_bf16.h>
#include <math.h>

typedef __attribute__((ext_vector_type(16))) __bf16 v16bf;
typedef __attribute__((ext_vector_type(8)))  float  v8f;

#define HIDDIM 128
#define NHEAD  4

// ---------------- utility device functions ----------------

__device__ inline float leakyrelu02(float x) { return x > 0.f ? x : 0.2f * x; }

__device__ inline void atomicMaxFloat(float* addr, float value) {
    // sign-aware monotonic mapping: works for mixed-sign floats
    if (value >= 0.f)
        atomicMax((int*)addr, __float_as_int(value));
    else
        atomicMin((unsigned int*)addr, __float_as_uint(value));
}

// ---------------- column stats (BatchNorm, training mode) ----------------
// one block per column, population variance (ddof=0) like jnp.var
__global__ void k_col_stats(const float* __restrict__ X, int n,
                            float* __restrict__ mean, float* __restrict__ rstd) {
    int c = blockIdx.x;                  // 0..127
    float s = 0.f, s2 = 0.f;
    for (int r = threadIdx.x; r < n; r += blockDim.x) {
        float v = X[(size_t)r * HIDDIM + c];
        s += v; s2 += v * v;
    }
    __shared__ float sh[256], sh2[256];
    sh[threadIdx.x] = s; sh2[threadIdx.x] = s2;
    __syncthreads();
    for (int off = 128; off > 0; off >>= 1) {
        if (threadIdx.x < off) {
            sh[threadIdx.x]  += sh[threadIdx.x + off];
            sh2[threadIdx.x] += sh2[threadIdx.x + off];
        }
        __syncthreads();
    }
    if (threadIdx.x == 0) {
        float m = sh[0] / (float)n;
        float v = sh2[0] / (float)n - m * m;
        mean[c] = m;
        rstd[c] = rsqrtf(v + 1e-5f);
    }
}

__global__ void k_bn_apply(const float* __restrict__ X, const float* __restrict__ mean,
                           const float* __restrict__ rstd, float* __restrict__ H, int total) {
    int i = blockIdx.x * blockDim.x + threadIdx.x;
    if (i >= total) return;
    int c = i & (HIDDIM - 1);
    H[i] = (X[i] - mean[c]) * rstd[c] + 1e-4f;
}

__global__ void k_bn_res_relu(const float* __restrict__ O, const float* __restrict__ mean,
                              const float* __restrict__ rstd, float* __restrict__ H, int total) {
    int i = blockIdx.x * blockDim.x + threadIdx.x;
    if (i >= total) return;
    int c = i & (HIDDIM - 1);
    float v = (O[i] - mean[c]) * rstd[c] + 1e-4f + H[i];
    H[i] = v > 0.f ? v : 0.f;
}

__global__ void k_relu_copy(const float* __restrict__ O, float* __restrict__ H, int total) {
    int i = blockIdx.x * blockDim.x + threadIdx.x;
    if (i >= total) return;
    float v = O[i];
    H[i] = v > 0.f ? v : 0.f;
}

// ---------------- WMMA GEMM: D = A(Nx128) @ W(128x128), bf16 in, f32 acc ----------------
// Block: 256 threads = 8 waves; each wave computes a 16-row x 128-col strip.
// W is staged into LDS PRE-PACKED in the exact per-lane B-fragment layout:
//   Wp[ct][k][lane] = 16 bf16 values (32B): one aligned 32B LDS load per operand.
// A is loaded branch-free as four float4 per k-step (pointer clamped, partial
// tiles masked by multiply) so the compiler emits unconditional global_load_b128.
__global__ void __launch_bounds__(256) k_gemm_wmma(
        const float* __restrict__ A, const float* __restrict__ W,
        float* __restrict__ Dst, int n) {
    __shared__ v16bf Wp[8 * 4 * 32];           // [ct][k][lane], 32KB, 32B-aligned
    __bf16* wps = (__bf16*)Wp;
    int tid = threadIdx.x;

    // stage + pack: W row-major element (kk, col) -> fragment position
    //   ct = col>>4, l16 = col&15, k = kk>>5, rem = kk&31
    //   lane = l16 + 16*((rem>>3)&1)        (rem 0-7,16-23 -> lo half; 8-15,24-31 -> hi half)
    //   elem = (rem&7) + 8*((rem>>4)&1)     (K pairs 0..7 in v0..3, 16..23 in v4..7)
    for (int i = tid; i < HIDDIM * HIDDIM; i += 256) {
        int kk  = i >> 7;
        int col = i & 127;
        int ct  = col >> 4, l16 = col & 15;
        int k   = kk >> 5,  rem = kk & 31;
        int lane = l16 + ((rem & 8) ? 16 : 0);
        int elem = (rem & 7) + ((rem & 16) ? 8 : 0);
        wps[((((ct << 2) + k) << 5) + lane) * 16 + elem] = (__bf16)W[i];
    }
    __syncthreads();

    int wave = tid >> 5;
    int lane = tid & 31;
    int l16  = lane & 15;
    int kb   = (lane < 16) ? 0 : 8;            // per ISA 16-bit A lane layout
    int rowbase = blockIdx.x * 128 + wave * 16;
    int row = rowbase + l16;
    bool rv = row < n;
    float scale = rv ? 1.f : 0.f;              // zero partial-tile rows by multiply
    const float4* a4 = (const float4*)(A + (size_t)(rv ? row : 0) * HIDDIM);

    v8f acc[8] = {};
#pragma unroll
    for (int k = 0; k < 4; k++) {
        int K0 = k * 32;
        // lane's A fragment: two contiguous 8-float runs, each = two float4
        int q = (K0 + kb) >> 2;                // kb in {0,8} -> float4-aligned
        float4 q0 = a4[q];
        float4 q1 = a4[q + 1];
        float4 q2 = a4[q + 4];                 // +16 floats
        float4 q3 = a4[q + 5];
        float t[16] = { q0.x, q0.y, q0.z, q0.w, q1.x, q1.y, q1.z, q1.w,
                        q2.x, q2.y, q2.z, q2.w, q3.x, q3.y, q3.z, q3.w };
        v16bf av;
#pragma unroll
        for (int e = 0; e < 16; e++) av[e] = (__bf16)(t[e] * scale);
#pragma unroll
        for (int ct = 0; ct < 8; ct++) {
            v16bf bv = Wp[((ct << 2) + k) * 32 + lane];   // one 32B LDS load
            acc[ct] = __builtin_amdgcn_wmma_f32_16x16x32_bf16(
                false, av, false, bv, (short)0, acc[ct], false, false);
        }
    }
    // C/D layout: VGPR r holds M=r (lanes 0-15) / M=8+r (lanes 16-31), N=lane&15
    int mbase = rowbase + ((lane < 16) ? 0 : 8);
#pragma unroll
    for (int ct = 0; ct < 8; ct++) {
#pragma unroll
        for (int r = 0; r < 8; r++) {
            int orow = mbase + r;
            if (orow < n)
                Dst[(size_t)orow * HIDDIM + ct * 16 + l16] = acc[ct][r];
        }
    }
}

// ---------------- attention scores a_s/a_d = <hW[n, h, :], att[h, :]> ----------------
__global__ void k_att_scores(const float* __restrict__ HW, const float* __restrict__ asv,
                             const float* __restrict__ adv, float* __restrict__ a_s,
                             float* __restrict__ a_d, int n) {
    int t = blockIdx.x * blockDim.x + threadIdx.x;
    if (t >= n * NHEAD) return;
    int node = t >> 2, hh = t & 3;
    const float* hrow = HW + (size_t)node * HIDDIM + hh * 32;
    const float* as_ = asv + hh * 32;
    const float* ad_ = adv + hh * 32;
    float ss = 0.f, sd = 0.f;
#pragma unroll
    for (int c = 0; c < 32; c++) { ss += hrow[c] * as_[c]; sd += hrow[c] * ad_[c]; }
    a_s[t] = ss; a_d[t] = sd;
}

// ---------------- init kernels ----------------
__global__ void k_init_out_bias(float* __restrict__ O, const float* __restrict__ b, int total) {
    int i = blockIdx.x * blockDim.x + threadIdx.x;
    if (i >= total) return;
    O[i] = b[i & (HIDDIM - 1)];        // fold GAT bias into accumulator init
}
__global__ void k_init_edge(float* __restrict__ amax, float* __restrict__ denom, int total) {
    int i = blockIdx.x * blockDim.x + threadIdx.x;
    if (i >= total) return;
    amax[i] = -1e30f; denom[i] = 0.f;
}
__global__ void k_zero(float* __restrict__ p, int total) {
    int i = blockIdx.x * blockDim.x + threadIdx.x;
    if (i >= total) return;
    p[i] = 0.f;
}

// ---------------- edge passes (softmax over incoming edges per dst) ----------------
__global__ void k_edge_amax(const int* __restrict__ ei, int E0, int Ep,
                            const float* __restrict__ a_s, const float* __restrict__ a_d,
                            float* __restrict__ amax) {
    int e = blockIdx.x * blockDim.x + threadIdx.x;
    if (e >= Ep) return;
    int s, d;
    if (e < E0) { s = ei[e]; d = ei[E0 + e]; } else { s = d = e - E0; }
#pragma unroll
    for (int h = 0; h < NHEAD; h++) {
        float a = leakyrelu02(a_s[s * NHEAD + h] + a_d[d * NHEAD + h]);
        atomicMaxFloat(&amax[d * NHEAD + h], a);
    }
}

__global__ void k_edge_denom(const int* __restrict__ ei, int E0, int Ep,
                             const float* __restrict__ a_s, const float* __restrict__ a_d,
                             const float* __restrict__ amax, float* __restrict__ denom) {
    int e = blockIdx.x * blockDim.x + threadIdx.x;
    if (e >= Ep) return;
    int s, d;
    if (e < E0) { s = ei[e]; d = ei[E0 + e]; } else { s = d = e - E0; }
#pragma unroll
    for (int h = 0; h < NHEAD; h++) {
        float a = leakyrelu02(a_s[s * NHEAD + h] + a_d[d * NHEAD + h]);
        atomicAdd(&denom[d * NHEAD + h], __expf(a - amax[d * NHEAD + h]));
    }
}

// one wave per edge; lane l handles channels [4l, 4l+4), head = l>>3
__global__ void k_edge_msg(const int* __restrict__ ei, int E0, int Ep,
                           const float* __restrict__ a_s, const float* __restrict__ a_d,
                           const float* __restrict__ amax, const float* __restrict__ denom,
                           const float* __restrict__ HW, float* __restrict__ O) {
    int e = blockIdx.x * 8 + (threadIdx.x >> 5);
    if (e >= Ep) return;
    int lane = threadIdx.x & 31;
    int s, d;
    if (e < E0) { s = ei[e]; d = ei[E0 + e]; } else { s = d = e - E0; }
    int h = lane >> 3;
    float a = leakyrelu02(a_s[s * NHEAD + h] + a_d[d * NHEAD + h]);
    float coef = __expf(a - amax[d * NHEAD + h]) / (denom[d * NHEAD + h] + 1e-16f);
    const float4 v = *(const float4*)(HW + (size_t)s * HIDDIM + lane * 4);
    float* ob = O + (size_t)d * HIDDIM + lane * 4;
    atomicAdd(ob + 0, v.x * coef);
    atomicAdd(ob + 1, v.y * coef);
    atomicAdd(ob + 2, v.z * coef);
    atomicAdd(ob + 3, v.w * coef);
}

// ---------------- pooling + FC + head ----------------
__global__ void k_pool(const float* __restrict__ H, const int* __restrict__ batch,
                       float* __restrict__ g, int total) {
    int i = blockIdx.x * blockDim.x + threadIdx.x;
    if (i >= total) return;
    int node = i >> 7, c = i & 127;
    atomicAdd(&g[(size_t)batch[node] * HIDDIM + c], H[i]);
}

__global__ void k_fc(const float* __restrict__ g, const float* __restrict__ mean,
                     const float* __restrict__ rstd, const float* __restrict__ fcw,
                     const float* __restrict__ fcb, float* __restrict__ g2, int total) {
    int i = blockIdx.x * blockDim.x + threadIdx.x;
    if (i >= total) return;
    int gi = i >> 7, j = i & 127;
    float acc = fcb[j];
    for (int k = 0; k < HIDDIM; k++) {
        float bn = (g[(size_t)gi * HIDDIM + k] - mean[k]) * rstd[k] + 1e-4f;
        acc += bn * fcw[k * HIDDIM + j];
    }
    g2[i] = acc > 0.f ? acc : 0.f;
}

__global__ void k_head(const float* __restrict__ g2, const float* __restrict__ mean,
                       const float* __restrict__ rstd, const float* __restrict__ cw,
                       const float* __restrict__ cb, float* __restrict__ outp, int Gn) {
    int gi = blockIdx.x * blockDim.x + threadIdx.x;
    if (gi >= Gn) return;
    float l0 = cb[0], l1 = cb[1];
    for (int k = 0; k < HIDDIM; k++) {
        float bn = (g2[(size_t)gi * HIDDIM + k] - mean[k]) * rstd[k] + 1e-4f;
        l0 += bn * cw[k * 2 + 0];
        l1 += bn * cw[k * 2 + 1];
    }
    float mx = fmaxf(l0, l1);
    float lse = mx + logf(__expf(l0 - mx) + __expf(l1 - mx));
    outp[gi * 2 + 0] = l0 - lse;
    outp[gi * 2 + 1] = l1 - lse;
}

// ---------------- host launch ----------------
extern "C" void kernel_launch(void* const* d_in, const int* in_sizes, int n_in,
                              void* d_out, int out_size, void* d_ws, size_t ws_size,
                              hipStream_t stream) {
    (void)n_in; (void)out_size; (void)ws_size;
    const float* x        = (const float*)d_in[0];
    const int*   ei       = (const int*)d_in[1];
    const int*   batch    = (const int*)d_in[2];
    const float* gat_lin  = (const float*)d_in[3];
    const float* att_src  = (const float*)d_in[4];
    const float* att_dst  = (const float*)d_in[5];
    const float* gat_bias = (const float*)d_in[6];
    const float* fc_w     = (const float*)d_in[7];
    const float* fc_b     = (const float*)d_in[8];
    const float* cls_w    = (const float*)d_in[9];
    const float* cls_b    = (const float*)d_in[10];
    float* outp = (float*)d_out;

    const int N  = in_sizes[0] / HIDDIM;
    const int E0 = in_sizes[1] / 2;
    const int Ep = E0 + N;
    const int NM = N * HIDDIM;
    const int N4 = N * NHEAD;
    const int Gn = 128;

    float* ws    = (float*)d_ws;
    float* h     = ws;
    float* hw    = h + NM;
    float* ob    = hw + NM;
    float* a_s   = ob + NM;
    float* a_d   = a_s + N4;
    float* amax  = a_d + N4;
    float* denom = amax + N4;
    float* mean  = denom + N4;
    float* rstd  = mean + HIDDIM;
    float* g     = rstd + HIDDIM;
    float* g2    = g + Gn * HIDDIM;
    float* mean2 = g2 + Gn * HIDDIM;
    float* rstd2 = mean2 + HIDDIM;
    float* mean3 = rstd2 + HIDDIM;
    float* rstd3 = mean3 + HIDDIM;

    dim3 blk(256);
    int gridNM   = (NM + 255) / 256;
    int gridE    = (Ep + 255) / 256;
    int gridEw   = (Ep + 7) / 8;        // one wave per edge, 8 waves/block
    int gridGemm = (N + 127) / 128;
    int gridN4   = (N4 + 255) / 256;
    int gridG    = (Gn * HIDDIM + 255) / 256;

    // input BN
    k_col_stats<<<HIDDIM, blk, 0, stream>>>(x, N, mean, rstd);
    k_bn_apply<<<gridNM, blk, 0, stream>>>(x, mean, rstd, h, NM);

    for (int l = 0; l < 4; l++) {
        const float* W  = gat_lin  + l * HIDDIM * HIDDIM;
        const float* as = att_src  + l * NHEAD * 32;
        const float* ad = att_dst  + l * NHEAD * 32;
        const float* bi = gat_bias + l * HIDDIM;

        k_gemm_wmma<<<gridGemm, blk, 0, stream>>>(h, W, hw, N);
        k_att_scores<<<gridN4, blk, 0, stream>>>(hw, as, ad, a_s, a_d, N);
        k_init_out_bias<<<gridNM, blk, 0, stream>>>(ob, bi, NM);
        k_init_edge<<<gridN4, blk, 0, stream>>>(amax, denom, N4);
        k_edge_amax<<<gridE, blk, 0, stream>>>(ei, E0, Ep, a_s, a_d, amax);
        k_edge_denom<<<gridE, blk, 0, stream>>>(ei, E0, Ep, a_s, a_d, amax, denom);
        k_edge_msg<<<gridEw, blk, 0, stream>>>(ei, E0, Ep, a_s, a_d, amax, denom, hw, ob);

        if (l == 0) {
            k_relu_copy<<<gridNM, blk, 0, stream>>>(ob, h, NM);
        } else {
            k_col_stats<<<HIDDIM, blk, 0, stream>>>(ob, N, mean, rstd);
            k_bn_res_relu<<<gridNM, blk, 0, stream>>>(ob, mean, rstd, h, NM);
        }
    }

    // global_add_pool -> BN -> FC -> BN -> classifier -> log_softmax
    k_zero<<<gridG, blk, 0, stream>>>(g, Gn * HIDDIM);
    k_pool<<<gridNM, blk, 0, stream>>>(h, batch, g, NM);
    k_col_stats<<<HIDDIM, blk, 0, stream>>>(g, Gn, mean2, rstd2);
    k_fc<<<gridG, blk, 0, stream>>>(g, mean2, rstd2, fc_w, fc_b, g2, Gn * HIDDIM);
    k_col_stats<<<HIDDIM, blk, 0, stream>>>(g2, Gn, mean3, rstd3);
    k_head<<<1, 128, 0, stream>>>(g2, mean3, rstd3, cls_w, cls_b, outp, Gn);
}
